// TopK_SparseScaledDotProductAttention_46325517255237
// MI455X (gfx1250) — matrix operand order repaired and checked by reading, not compiled, not run on hardware
//
#include <hip/hip_runtime.h>

typedef __attribute__((ext_vector_type(16))) _Float16 v16h;
typedef __attribute__((ext_vector_type(8)))  _Float16 v8h;
typedef __attribute__((ext_vector_type(4)))  _Float16 v4h;
typedef __attribute__((ext_vector_type(8)))  float    v8f;
typedef __attribute__((ext_vector_type(4)))  float    v4f;

#define S_LEN 1024
#define D_DIM 64
#define TOPK  32

__device__ __forceinline__ unsigned f2key(float f) {
  unsigned u = __float_as_uint(f);
  return u ^ ((u & 0x80000000u) ? 0xFFFFFFFFu : 0x80000000u);
}
__device__ __forceinline__ float key2f(unsigned k) {
  unsigned u = (k & 0x80000000u) ? (k ^ 0x80000000u) : ~k;
  return __uint_as_float(u);
}

typedef union { v16h v; v8h h[2]; } u16h;

__global__ __launch_bounds__(256)
void topk_attn_kernel(const float* __restrict__ Q,
                      const float* __restrict__ K,
                      const float* __restrict__ V,
                      float* __restrict__ ctx,
                      float* __restrict__ attn)
{
  __shared__ __align__(16) float    sc[16 * S_LEN];   // 64 KB f32 scores
  __shared__ __align__(16) _Float16 un[16 * S_LEN];   // 32 KB f16 unnormalized
  __shared__ float s_th[16];
  __shared__ float s_inv[16];

  const int blk   = blockIdx.x;
  const int qt    = blk & 63;             // 64 query tiles of 16
  const int bh    = blk >> 6;             // 128 (b,h) pairs
  const int tid   = threadIdx.x;
  const int wave  = tid >> 5;
  const int lane  = tid & 31;
  const int lhalf = lane >> 4;            // 0: lanes 0-15, 1: lanes 16-31
  const int l16   = lane & 15;

  const size_t baseQKV = (size_t)bh * (S_LEN * D_DIM);
  const int qrow0 = qt * 16;

  // ---------------- Phase A: scores = (Q Ktile^T) / 8 via WMMA f16 ----------
  // A layout (16x32 f16): lane = row M=l16; halves 0..7 -> K = lhalf*16+h,
  // halves 8..15 -> K = lhalf*16 + 8 + h.  a_lo covers D 0..31, a_hi 32..63.
  const float* qrowp = Q + baseQKV + (size_t)(qrow0 + l16) * D_DIM;
  v16h a_lo = {}, a_hi = {};
  #pragma unroll
  for (int h = 0; h < 8; ++h) {
    a_lo[h]     = (_Float16)qrowp[     lhalf * 16 + h];
    a_lo[h + 8] = (_Float16)qrowp[ 8 + lhalf * 16 + h];
    a_hi[h]     = (_Float16)qrowp[32 + lhalf * 16 + h];
    a_hi[h + 8] = (_Float16)qrowp[40 + lhalf * 16 + h];
  }

  #pragma unroll 1
  for (int t = 0; t < 8; ++t) {
    const int n0 = wave * 128 + t * 16;
    // B layout (32x16 f16): lane = col N=l16; half h -> K = lhalf*16 + h.
    // B[k][n] = K[n0+n][d0+k]  (K^T tile): contiguous 16-float run per lane.
    const float* krowp = K + baseQKV + (size_t)(n0 + l16) * D_DIM + lhalf * 16;
    v16h b_lo = {}, b_hi = {};
    #pragma unroll
    for (int h = 0; h < 16; ++h) {
      b_lo[h] = (_Float16)krowp[h];        // D 0..31
      b_hi[h] = (_Float16)krowp[32 + h];   // D 32..63
    }
    v8f c = {};
    c = __builtin_amdgcn_wmma_f32_16x16x32_f16(false, a_lo, false, b_lo,
                                               (short)0, c, false, false);
    c = __builtin_amdgcn_wmma_f32_16x16x32_f16(false, a_hi, false, b_hi,
                                               (short)0, c, false, false);
    // C layout: VGPR v -> M = v + lhalf*8, N = l16
    #pragma unroll
    for (int v = 0; v < 8; ++v)
      sc[(v + lhalf * 8) * S_LEN + n0 + l16] = c[v] * 0.125f;
  }
  __syncthreads();

  // ------------- Phase B: per-row 32nd-largest threshold + denominator ------
  #pragma unroll 1
  for (int rr = 0; rr < 2; ++rr) {
    const int row = wave * 2 + rr;
    float    vals[32];
    unsigned keys[32];
    #pragma unroll
    for (int j = 0; j < 32; ++j) {
      vals[j] = sc[row * S_LEN + j * 32 + lane];
      keys[j] = f2key(vals[j]);
    }
    // Radix binary search: largest x with count(key >= x) >= TOPK == k-th key
    unsigned prefix = 0u;
    #pragma unroll 1
    for (int bit = 31; bit >= 0; --bit) {
      const unsigned trial = prefix | (1u << bit);
      int cnt = 0;
      #pragma unroll
      for (int j = 0; j < 32; ++j) cnt += (keys[j] >= trial) ? 1 : 0;
      #pragma unroll
      for (int off = 16; off > 0; off >>= 1) cnt += __shfl_xor(cnt, off, 32);
      if (cnt >= TOPK) prefix = trial;
    }
    // denominator: selected -> exp(s), others -> exp(0)=1
    float part = 0.0f;
    #pragma unroll
    for (int j = 0; j < 32; ++j)
      part += (keys[j] >= prefix) ? __expf(vals[j]) : 1.0f;
    #pragma unroll
    for (int off = 16; off > 0; off >>= 1) part += __shfl_xor(part, off, 32);
    if (lane == 0) {
      s_th[row]  = key2f(prefix);
      s_inv[row] = 1.0f / (part + 1e-8f);
    }
  }
  __syncthreads();

  // ------------- Phase C: stream attn to HBM (NT) + stash f16 unnormalized --
  {
    const size_t attnBase = ((size_t)bh * S_LEN + qrow0) * S_LEN;
    #pragma unroll 1
    for (int row = 0; row < 16; ++row) {
      const float th  = s_th[row];
      const float inv = s_inv[row];
      const int   col = tid * 4;                 // 256 threads x float4 = row
      const v4f s4 = *(const v4f*)&sc[row * S_LEN + col];
      const float u0 = (s4.x >= th) ? __expf(s4.x) : 1.0f;
      const float u1 = (s4.y >= th) ? __expf(s4.y) : 1.0f;
      const float u2 = (s4.z >= th) ? __expf(s4.z) : 1.0f;
      const float u3 = (s4.w >= th) ? __expf(s4.w) : 1.0f;
      v4f a4;
      a4.x = u0 * inv; a4.y = u1 * inv; a4.z = u2 * inv; a4.w = u3 * inv;
      // attn is written once, never read: non-temporal keeps K/V hot in L2
      __builtin_nontemporal_store(
          a4, (v4f*)(attn + attnBase + (size_t)row * S_LEN + col));
      v4h h4;
      h4[0] = (_Float16)u0; h4[1] = (_Float16)u1;
      h4[2] = (_Float16)u2; h4[3] = (_Float16)u3;
      *(v4h*)&un[row * S_LEN + col] = h4;
    }
  }
  __syncthreads();

  // ------------- Phase D: context = (unnorm @ V) * inv via WMMA f16 ---------
  if (wave < 4) {
    const int n0 = wave * 16;                    // output column tile of D=64
    const float* vbase = V + baseQKV + n0 + l16; // fixed column per lane
    v8f c = {};
    #pragma unroll 1
    for (int ks = 0; ks < 32; ++ks) {
      const int k0 = ks * 32;
      u16h a;                                    // A: unnorm 16x32 tile (LDS)
      a.h[0] = *(const v8h*)&un[l16 * S_LEN + k0 +     lhalf * 16];
      a.h[1] = *(const v8h*)&un[l16 * S_LEN + k0 + 8 + lhalf * 16];
      v16h b = {};                               // B: V 32x16 tile (global)
      const float* vp = vbase + (size_t)(k0 + lhalf * 16) * D_DIM;
      #pragma unroll
      for (int h = 0; h < 16; ++h)
        b[h] = (_Float16)vp[(size_t)h * D_DIM]; // lanes coalesce per h
      c = __builtin_amdgcn_wmma_f32_16x16x32_f16(false, a.v, false, b,
                                                 (short)0, c, false, false);
    }
    const size_t ctxBase = ((size_t)bh * S_LEN + qrow0) * D_DIM;
    #pragma unroll
    for (int v = 0; v < 8; ++v) {
      const int m = v + lhalf * 8;
      __builtin_nontemporal_store(
          c[v] * s_inv[m],
          ctx + ctxBase + (size_t)m * D_DIM + n0 + l16);
    }
  }
}

extern "C" void kernel_launch(void* const* d_in, const int* in_sizes, int n_in,
                              void* d_out, int out_size, void* d_ws, size_t ws_size,
                              hipStream_t stream) {
  (void)in_sizes; (void)n_in; (void)out_size; (void)d_ws; (void)ws_size;
  const float* Q = (const float*)d_in[0];
  const float* K = (const float*)d_in[1];
  const float* V = (const float*)d_in[2];
  float* ctx  = (float*)d_out;
  float* attn = ctx + (size_t)8 * 16 * 1024 * 64;   // context first, then attn
  dim3 grid(128 * 64);                              // (b*h) x (S/16) tiles
  topk_attn_kernel<<<grid, dim3(256), 0, stream>>>(Q, K, V, ctx, attn);
}
